// GruAgent_33517924778626
// MI455X (gfx1250) — matrix operand order repaired and checked by reading, not compiled, not run on hardware
//
#include <hip/hip_runtime.h>

// ---------------------------------------------------------------------------
// GRU agent forward for MI455X (gfx1250, wave32, WMMA 16x16x32 f16).
//   T=512, B=256, OBS=128, H=64, A=16
// Kernel 1: sequential GRU scan (16 blocks x 4 waves; wave w owns h-cols
//           16w..16w+15 == its own r/z/n gate tiles). Hidden state is
//           sign-quantized (exact in f16) -> recurrent GEMM in f16 WMMA.
// Kernel 2: fused 192->64->64->{16,1} MLP per 16-row tile, f16 WMMA, weights
//           staged once in LDS, activations restaged via LDS + s_wait_dscnt.
// ---------------------------------------------------------------------------

typedef __attribute__((ext_vector_type(16))) _Float16 v16h;
typedef __attribute__((ext_vector_type(8)))  float    v8f;

#define TT   512
#define BB   256
#define OBSD 128
#define HD   64
#define ROWS (TT * BB)        // 131072
#define AD   16

__device__ __forceinline__ v8f wmma16(v16h a, v16h b, v8f c) {
  // D = A(16x32 f16) * B(32x16 f16) + C(16x16 f32)
  return __builtin_amdgcn_wmma_f32_16x16x32_f16(
      /*neg_a=*/false, a, /*neg_b=*/false, b,
      /*c_mod=*/(short)0, c, /*reuse_a=*/false, /*reuse_b=*/false);
}

__device__ __forceinline__ float sigmoidf_(float x) {
  return 1.0f / (1.0f + __expf(-x));
}
__device__ __forceinline__ float signf_(float x) {
  return (x > 0.0f) ? 1.0f : ((x < 0.0f) ? -1.0f : 0.0f);
}
__device__ __forceinline__ void lds_wait() {
  // make this wave's LDS stores visible to all its lanes before cross-lane reads
  asm volatile("s_wait_dscnt 0" ::: "memory");
}

// ---------------------------------------------------------------------------
// Kernel 1: GRU scan.  grid = B/16 = 16 blocks, block = 128 threads (4 waves).
// ---------------------------------------------------------------------------
__global__ __launch_bounds__(128) void gru_scan_kernel(
    const float* __restrict__ x,        // [T*B, OBS]
    const float* __restrict__ h0,       // [B, H]
    const float* __restrict__ done,     // [T*B]
    const float* __restrict__ w_ih,     // [3H, OBS]
    const float* __restrict__ w_hh,     // [3H, H]
    const float* __restrict__ b_ih,     // [3H]
    const float* __restrict__ b_hh,     // [3H]
    _Float16*    __restrict__ hidden16, // [T*B, H] (sign values, f16)
    float*       __restrict__ out_state)// [B, H]
{
  __shared__ _Float16 hsh[16][HD];      // masked hidden tile published per step

  const int tid  = threadIdx.x;
  const int w    = tid >> 5;            // wave 0..3
  const int lane = tid & 31;
  const int li   = lane & 15;
  const int hi   = lane >> 4;           // lane half
  const int rbase = blockIdx.x * 16;    // batch-row base
  const int cR   = w * 16 + li;         // this lane's h / gate column (0..63)

  // --- hoist B fragments (f16) into registers: gi (K=128 -> 4 tiles),
  //     gh (K=64 -> 2 tiles); gates r,z,n are N-tiles w, w+4, w+8.
  v16h BgiR[4], BgiZ[4], BgiN[4], BhhR[2], BhhZ[2], BhhN[2];
#pragma unroll
  for (int k = 0; k < 4; ++k) {
    const int k0 = 32 * k + (hi ? 16 : 0);
    const float* wr = w_ih + (size_t)(      cR) * OBSD + k0;
    const float* wz = w_ih + (size_t)( 64 + cR) * OBSD + k0;
    const float* wn = w_ih + (size_t)(128 + cR) * OBSD + k0;
    v16h fr, fz, fn;
#pragma unroll
    for (int j = 0; j < 16; ++j) {
      fr[j] = (_Float16)wr[j]; fz[j] = (_Float16)wz[j]; fn[j] = (_Float16)wn[j];
    }
    BgiR[k] = fr; BgiZ[k] = fz; BgiN[k] = fn;
  }
#pragma unroll
  for (int k = 0; k < 2; ++k) {
    const int k0 = 32 * k + (hi ? 16 : 0);
    const float* wr = w_hh + (size_t)(      cR) * HD + k0;
    const float* wz = w_hh + (size_t)( 64 + cR) * HD + k0;
    const float* wn = w_hh + (size_t)(128 + cR) * HD + k0;
    v16h fr, fz, fn;
#pragma unroll
    for (int j = 0; j < 16; ++j) {
      fr[j] = (_Float16)wr[j]; fz[j] = (_Float16)wz[j]; fn[j] = (_Float16)wn[j];
    }
    BhhR[k] = fr; BhhZ[k] = fz; BhhN[k] = fn;
  }

  const float biR  = b_ih[cR]       + b_hh[cR];
  const float biZ  = b_ih[64 + cR]  + b_hh[64 + cR];
  const float biNI = b_ih[128 + cR];
  const float biNH = b_hh[128 + cR];

  // initial hidden (raw f32 state; reference uses h0 unquantized)
  float hprev[8];
#pragma unroll
  for (int i = 0; i < 8; ++i) {
    const int row = i + 8 * hi;
    hprev[i] = h0[(size_t)(rbase + row) * HD + cR];
  }

  for (int t = 0; t < TT; ++t) {
    const int xrow0 = t * BB + rbase;

    // 1) done-mask carried state, publish f16 to LDS for gh A-fragments
    float hm[8];
#pragma unroll
    for (int i = 0; i < 8; ++i) {
      const int row = i + 8 * hi;
      const float d = done[xrow0 + row];
      hm[i] = (1.0f - d) * hprev[i];
      hsh[row][cR] = (_Float16)hm[i];
    }
    __syncthreads();

    v8f accR = {}, accZ = {}, accNI = {}, accNH = {};

    // 2) gh = h_masked @ w_hh^T  (K = 64 -> 2 WMMA k-steps)
#pragma unroll
    for (int k = 0; k < 2; ++k) {
      v16h A;
      const int kb = 32 * k + (hi ? 16 : 0);
#pragma unroll
      for (int j = 0; j < 8; ++j) {
        A[j]     = hsh[li][kb + j];
        A[8 + j] = hsh[li][kb + 8 + j];
      }
      accR  = wmma16(A, BhhR[k], accR);
      accZ  = wmma16(A, BhhZ[k], accZ);
      accNH = wmma16(A, BhhN[k], accNH);
    }

    // 3) gi = x @ w_ih^T  (K = 128 -> 4 WMMA k-steps), x converted f32->f16
#pragma unroll
    for (int k = 0; k < 4; ++k) {
      v16h A;
      const int kb = 32 * k + (hi ? 16 : 0);
      const float* xp = x + (size_t)(xrow0 + li) * OBSD;
#pragma unroll
      for (int j = 0; j < 8; ++j) {
        A[j]     = (_Float16)xp[kb + j];
        A[8 + j] = (_Float16)xp[kb + 8 + j];
      }
      accR  = wmma16(A, BgiR[k], accR);
      accZ  = wmma16(A, BgiZ[k], accZ);
      accNI = wmma16(A, BgiN[k], accNI);
    }

    // prefetch next step's x tile (global_prefetch_b8)
    if (t + 1 < TT) __builtin_prefetch(x + (size_t)(xrow0 + BB + li) * OBSD, 0, 1);

    // 4) gates + STE-sign; store hidden, keep carried state in registers
#pragma unroll
    for (int i = 0; i < 8; ++i) {
      const float r  = sigmoidf_(accR[i] + biR);
      const float z  = sigmoidf_(accZ[i] + biZ);
      const float n  = tanhf(accNI[i] + biNI + r * (accNH[i] + biNH));
      const float hn = (1.0f - z) * n + z * hm[i];
      const float s  = signf_(hn);
      hprev[i] = s;
      const int row = i + 8 * hi;
      hidden16[(size_t)(xrow0 + row) * HD + cR] = (_Float16)s;
    }
    __syncthreads();  // protect hsh (read above) from next iteration's writes
  }

#pragma unroll
  for (int i = 0; i < 8; ++i) {
    const int row = i + 8 * hi;
    out_state[(size_t)(rbase + row) * HD + cR] = hprev[i];
  }
}

// ---------------------------------------------------------------------------
// Kernel 2: fused MLP (192 -> 64 -> 64 -> n3), n3 = 16 (actor) or 1 (critic).
// grid = 512 blocks x 128 threads (4 waves); 2048 waves x 4 tiles = 8192 tiles.
// ---------------------------------------------------------------------------
__global__ __launch_bounds__(128) void mlp_kernel(
    const _Float16* __restrict__ hidden16, // [T*B, H]
    const float*    __restrict__ x,        // [T*B, OBS]
    const float* __restrict__ w1, const float* __restrict__ b1,   // [64,192],[64]
    const float* __restrict__ w2, const float* __restrict__ b2,   // [64,64],[64]
    const float* __restrict__ w3, const float* __restrict__ b3,   // [n3,64],[n3]
    float* __restrict__ out, int n3)
{
  __shared__ _Float16 w1h[64][192];   // 24 KB  (B^T rows contiguous in K)
  __shared__ _Float16 w2h[64][64];    //  8 KB
  __shared__ _Float16 w3h[16][64];    //  2 KB  (rows >= n3 zeroed)
  __shared__ _Float16 yb[4][16][HD];  //  8 KB  per-wave activation restage

  const int tid  = threadIdx.x;
  const int w    = tid >> 5;
  const int lane = tid & 31;
  const int li   = lane & 15;
  const int hi   = lane >> 4;

  for (int idx = tid; idx < 64 * 192; idx += 128)
    w1h[idx / 192][idx % 192] = (_Float16)w1[idx];
  for (int idx = tid; idx < 64 * 64; idx += 128)
    w2h[idx >> 6][idx & 63] = (_Float16)w2[idx];
  for (int idx = tid; idx < 16 * 64; idx += 128) {
    const int rr = idx >> 6, cc = idx & 63;
    w3h[rr][cc] = (rr < n3) ? (_Float16)w3[rr * 64 + cc] : (_Float16)0.0f;
  }
  __syncthreads();

  const int nwaves = gridDim.x * 4;
  for (int tile = blockIdx.x * 4 + w; tile < ROWS / 16; tile += nwaves) {
    const size_t r0 = (size_t)tile * 16;

    // ---- L1: cat[16,192] @ w1^T -> [16,64] ------------------------------
    v8f a1[4];
#pragma unroll
    for (int n = 0; n < 4; ++n) { v8f z = {}; a1[n] = z; }
#pragma unroll
    for (int kt = 0; kt < 6; ++kt) {
      v16h A;
      if (kt < 2) {                       // hidden part (already f16 signs)
        const int kb = 32 * kt + (hi ? 16 : 0);
        const _Float16* hp = hidden16 + (r0 + li) * HD;
#pragma unroll
        for (int j = 0; j < 8; ++j) { A[j] = hp[kb + j]; A[8 + j] = hp[kb + 8 + j]; }
      } else {                            // x part, convert f32 -> f16
        const int kb = 32 * (kt - 2) + (hi ? 16 : 0);
        const float* xp = x + (r0 + li) * OBSD;
#pragma unroll
        for (int j = 0; j < 8; ++j) {
          A[j]     = (_Float16)xp[kb + j];
          A[8 + j] = (_Float16)xp[kb + 8 + j];
        }
      }
      const int k0 = 32 * kt + (hi ? 16 : 0);
#pragma unroll
      for (int n = 0; n < 4; ++n) {
        v16h Bv;
        const _Float16* wp = &w1h[n * 16 + li][k0];
#pragma unroll
        for (int j = 0; j < 16; ++j) Bv[j] = wp[j];
        a1[n] = wmma16(A, Bv, a1[n]);
      }
    }
#pragma unroll
    for (int n = 0; n < 4; ++n) {
      const float bb = b1[n * 16 + li];
#pragma unroll
      for (int i = 0; i < 8; ++i)
        yb[w][i + 8 * hi][n * 16 + li] = (_Float16)tanhf(a1[n][i] + bb);
    }
    lds_wait();

    // ---- L2: [16,64] @ w2^T -> [16,64] ----------------------------------
    v8f a2[4];
#pragma unroll
    for (int n = 0; n < 4; ++n) { v8f z = {}; a2[n] = z; }
#pragma unroll
    for (int kt = 0; kt < 2; ++kt) {
      v16h A;
      const int kb = 32 * kt + (hi ? 16 : 0);
#pragma unroll
      for (int j = 0; j < 8; ++j) { A[j] = yb[w][li][kb + j]; A[8 + j] = yb[w][li][kb + 8 + j]; }
      const int k0 = 32 * kt + (hi ? 16 : 0);
#pragma unroll
      for (int n = 0; n < 4; ++n) {
        v16h Bv;
        const _Float16* wp = &w2h[n * 16 + li][k0];
#pragma unroll
        for (int j = 0; j < 16; ++j) Bv[j] = wp[j];
        a2[n] = wmma16(A, Bv, a2[n]);
      }
    }
#pragma unroll
    for (int n = 0; n < 4; ++n) {
      const float bb = b2[n * 16 + li];
#pragma unroll
      for (int i = 0; i < 8; ++i)
        yb[w][i + 8 * hi][n * 16 + li] = (_Float16)tanhf(a2[n][i] + bb);
    }
    lds_wait();

    // ---- L3: [16,64] @ w3^T -> [16,n3] ----------------------------------
    v8f a3 = {};
#pragma unroll
    for (int kt = 0; kt < 2; ++kt) {
      v16h A;
      const int kb = 32 * kt + (hi ? 16 : 0);
#pragma unroll
      for (int j = 0; j < 8; ++j) { A[j] = yb[w][li][kb + j]; A[8 + j] = yb[w][li][kb + 8 + j]; }
      v16h Bv;
      const _Float16* wp = &w3h[li][32 * kt + (hi ? 16 : 0)];
#pragma unroll
      for (int j = 0; j < 16; ++j) Bv[j] = wp[j];
      a3 = wmma16(A, Bv, a3);
    }
    if (n3 == AD) {
      const float bb = b3[li];
#pragma unroll
      for (int i = 0; i < 8; ++i)
        out[(r0 + i + 8 * hi) * AD + li] = a3[i] + bb;
    } else {                               // critic: only column 0 is real
      if (li == 0) {
        const float bb = b3[0];
#pragma unroll
        for (int i = 0; i < 8; ++i) out[r0 + i + 8 * hi] = a3[i] + bb;
      }
    }
  }
}

// ---------------------------------------------------------------------------
extern "C" void kernel_launch(void* const* d_in, const int* in_sizes, int n_in,
                              void* d_out, int out_size, void* d_ws, size_t ws_size,
                              hipStream_t stream) {
  (void)in_sizes; (void)n_in; (void)out_size; (void)ws_size;
  const float* x    = (const float*)d_in[0];
  const float* h0   = (const float*)d_in[1];
  const float* done = (const float*)d_in[2];
  const float* w_ih = (const float*)d_in[3];
  const float* w_hh = (const float*)d_in[4];
  const float* b_ih = (const float*)d_in[5];
  const float* b_hh = (const float*)d_in[6];
  const float* a_w1 = (const float*)d_in[7];
  const float* a_b1 = (const float*)d_in[8];
  const float* a_w2 = (const float*)d_in[9];
  const float* a_b2 = (const float*)d_in[10];
  const float* a_w3 = (const float*)d_in[11];
  const float* a_b3 = (const float*)d_in[12];
  const float* c_w1 = (const float*)d_in[13];
  const float* c_b1 = (const float*)d_in[14];
  const float* c_w2 = (const float*)d_in[15];
  const float* c_b2 = (const float*)d_in[16];
  const float* c_w3 = (const float*)d_in[17];
  const float* c_b3 = (const float*)d_in[18];

  float* logits    = (float*)d_out;                    // [131072, 16]
  float* value     = logits + (size_t)ROWS * AD;       // [131072]
  float* new_state = value + ROWS;                     // [256, 64]
  _Float16* hidden16 = (_Float16*)d_ws;                // [131072, 64] f16

  hipLaunchKernelGGL(gru_scan_kernel, dim3(BB / 16), dim3(128), 0, stream,
                     x, h0, done, w_ih, w_hh, b_ih, b_hh, hidden16, new_state);
  hipLaunchKernelGGL(mlp_kernel, dim3(512), dim3(128), 0, stream,
                     hidden16, x, a_w1, a_b1, a_w2, a_b2, a_w3, a_b3, logits, AD);
  hipLaunchKernelGGL(mlp_kernel, dim3(512), dim3(128), 0, stream,
                     hidden16, x, c_w1, c_b1, c_w2, c_b2, c_w3, c_b3, value, 1);
}
